// OptimizedRWKVBlock_39745627357523
// MI455X (gfx1250) — matrix operand order repaired and checked by reading, not compiled, not run on hardware
//
#include <hip/hip_runtime.h>

// ---------------------------------------------------------------------------
// RWKV block for MI455X (gfx1250).
//  - bf16 WMMA GEMMs (f32 accum), 128x128 block tile, 32x64 wave tile
//  - A tiles: double-buffered GLOBAL_LOAD_ASYNC_TO_LDS_B128 (ASYNCcnt)
//  - B tiles: Tensor Data Mover (tensor_load_to_lds, TENSORcnt) with HW LDS
//    padding reproducing the bank-spread 40-ushort row stride
//  - weights pre-transposed to [N,K] bf16 so all tile copies are contiguous
//  - fused LN/shift/mix, fused sigmoid / relu^2 epilogues
//  - WKV scan with 2 independent chains per thread for TRANS-latency hiding
// Shapes fixed by reference: B=8, T=2048, C=1024, H=4096.
// ---------------------------------------------------------------------------

#define LN_EPS 1e-5f
#define NEG_INF -1e38f
#define LDK 40  // padded LDS row stride (ushorts): 80B = 16B-aligned, bank-spread

#if __has_builtin(__builtin_amdgcn_tensor_load_to_lds)
#define HAVE_TDM 1
#else
#define HAVE_TDM 0
#endif

typedef __attribute__((ext_vector_type(16))) __bf16 v16bf;
typedef __attribute__((ext_vector_type(8)))  float  v8f;
typedef __attribute__((ext_vector_type(4))) unsigned int u32x4;
typedef __attribute__((ext_vector_type(8))) int          i32x8;
typedef __attribute__((ext_vector_type(4))) int          i32x4;

union FragBF { v16bf v; unsigned u[8]; };
union FragF  { v8f  v; float    f[8]; };

__device__ __forceinline__ unsigned short f2bf(float f) {
  unsigned u = __float_as_uint(f);
  return (unsigned short)((u + 0x7FFFu + ((u >> 16) & 1u)) >> 16);  // RNE
}

__device__ __forceinline__ float wave_sum(float v) {
#pragma unroll
  for (int off = 16; off > 0; off >>= 1) v += __shfl_xor(v, off, 32);
  return v;
}

__device__ __forceinline__ float block_sum256(float v, float* sbuf) {
  int lane = threadIdx.x & 31, w = threadIdx.x >> 5;
  v = wave_sum(v);
  __syncthreads();
  if (lane == 0) sbuf[w] = v;
  __syncthreads();
  float r = sbuf[0];
#pragma unroll
  for (int i = 1; i < 8; ++i) r += sbuf[i];
  return r;
}

// ---------------------------------------------------------------------------
// Async 16B copy global -> LDS (per lane). Tracked by ASYNCcnt.
// ---------------------------------------------------------------------------
__device__ __forceinline__ void async_copy16(const unsigned short* g,
                                             unsigned short* l) {
  unsigned lds_addr = (unsigned)(unsigned long long)l;  // low 32b = LDS offset
  unsigned long long gaddr = (unsigned long long)g;
  asm volatile("global_load_async_to_lds_b128 %0, %1, off"
               :
               : "v"(lds_addr), "v"(gaddr)
               : "memory");
}

__device__ __forceinline__ void wait_async0() {
  asm volatile("s_wait_asynccnt 0" ::: "memory");
}

// ---------------------------------------------------------------------------
// TDM 2D tile load: tile_x elements (2B each) x tile_y rows, global row
// stride = stride_elems, into LDS at lds_byte_addr with HW padding of
// 4 DWORDs after every 16 DWORDs (=> 40-ushort LDS row stride).
// ---------------------------------------------------------------------------
#if HAVE_TDM
__device__ __forceinline__ void tdm_load_2d(unsigned lds_byte_addr,
                                            const void* gaddr, int tile_x,
                                            int tile_y, int stride_elems,
                                            int tensor_x, int tensor_y) {
  unsigned long long ga = (unsigned long long)gaddr;
  u32x4 g0;
  g0[0] = 1u;                                    // count=1 (valid user D#)
  g0[1] = lds_byte_addr;                         // lds_addr
  g0[2] = (unsigned)(ga & 0xFFFFFFFFu);          // global_addr[31:0]
  g0[3] = (unsigned)((ga >> 32) & 0x01FFFFFFu)   // global_addr[56:32]
          | (2u << 30);                          // type = 2 ("image")
  i32x8 g1;
  g1[0] = (1 << 16)      // data_size = 2 bytes
        | (1 << 20)      // pad_enable
        | (3 << 22)      // pad_interval: 16 DWORDs
        | (3 << 25);     // pad_amount: 4 DWORDs
  g1[1] = (int)((unsigned)tensor_x << 16);                       // dim0 lo16
  g1[2] = (int)((((unsigned)tensor_x >> 16) & 0xFFFFu)
                | ((unsigned)tensor_y << 16));                   // dim0 hi/dim1 lo
  g1[3] = (int)((((unsigned)tensor_y >> 16) & 0xFFFFu)
                | ((unsigned)tile_x << 16));                     // dim1 hi/tile0
  g1[4] = tile_y & 0xFFFF;                                       // tile1 (tile2=0)
  g1[5] = stride_elems;                                          // dim0_stride lo
  g1[6] = 0;
  g1[7] = 0;
  i32x4 z4 = {0, 0, 0, 0};
  i32x8 z8 = {0, 0, 0, 0, 0, 0, 0, 0};
  __builtin_amdgcn_tensor_load_to_lds(g0, g1, z4, z4, z8, 0);
}
#endif

__device__ __forceinline__ v8f wmma_bf16(v16bf a, v16bf b, v8f c) {
  return __builtin_amdgcn_wmma_f32_16x16x32_bf16(false, a, false, b, (short)0,
                                                 c, false, false);
}

// ---------------------------------------------------------------------------
// Weight conversion with transpose: in [K,N] f32 -> out [N,K] bf16.
// ---------------------------------------------------------------------------
__global__ __launch_bounds__(256) void k_convert_t(
    const float* __restrict__ in, unsigned short* __restrict__ out, int K,
    int N) {
  __shared__ unsigned short tile[32][33];
  int k0 = blockIdx.y * 32, n0 = blockIdx.x * 32;
  int tx = threadIdx.x & 31, ty = threadIdx.x >> 5;  // 32 x 8
#pragma unroll
  for (int i = 0; i < 4; ++i) {
    int kl = ty + i * 8;
    tile[kl][tx] = f2bf(in[(size_t)(k0 + kl) * N + n0 + tx]);
  }
  __syncthreads();
#pragma unroll
  for (int i = 0; i < 4; ++i) {
    int nl = ty + i * 8;
    out[(size_t)(n0 + nl) * K + k0 + tx] = tile[tx][nl];
  }
}

// ---------------------------------------------------------------------------
// Fused LayerNorm + time-shift + token-mix -> bf16 activations.
// One 256-thread block per row (b,t); C == 1024 (4 channels/thread).
// ---------------------------------------------------------------------------
template <int NOUT>
__global__ __launch_bounds__(256) void k_ln_shift_mix(
    const float* __restrict__ x, const float* __restrict__ g,
    const float* __restrict__ b, const float* __restrict__ mx0,
    const float* __restrict__ mx1, const float* __restrict__ mx2,
    unsigned short* __restrict__ o0, unsigned short* __restrict__ o1,
    unsigned short* __restrict__ o2, int T, int C) {
  __shared__ float sbuf[8];
  int m = blockIdx.x;
  int t = m % T;  // shift is per-batch along T
  int c0 = threadIdx.x * 4;
  const float* row = x + (size_t)m * C;

  float4 xv = *(const float4*)(row + c0);
  float mu, rs;
  {
    float s = block_sum256(xv.x + xv.y + xv.z + xv.w, sbuf);
    mu = s / C;
    float d0 = xv.x - mu, d1 = xv.y - mu, d2 = xv.z - mu, d3 = xv.w - mu;
    float q = block_sum256(d0 * d0 + d1 * d1 + d2 * d2 + d3 * d3, sbuf);
    rs = rsqrtf(q / C + LN_EPS);
  }
  float4 gv = *(const float4*)(g + c0);
  float4 bv = *(const float4*)(b + c0);
  float h0 = (xv.x - mu) * rs * gv.x + bv.x;
  float h1 = (xv.y - mu) * rs * gv.y + bv.y;
  float h2 = (xv.z - mu) * rs * gv.z + bv.z;
  float h3 = (xv.w - mu) * rs * gv.w + bv.w;

  float p0 = 0.f, p1 = 0.f, p2 = 0.f, p3 = 0.f;
  if (t > 0) {  // uniform per block -> barriers inside are safe
    float4 pv = *(const float4*)(row - C + c0);
    float mu2, rs2;
    float s = block_sum256(pv.x + pv.y + pv.z + pv.w, sbuf);
    mu2 = s / C;
    float d0 = pv.x - mu2, d1 = pv.y - mu2, d2 = pv.z - mu2, d3 = pv.w - mu2;
    float q = block_sum256(d0 * d0 + d1 * d1 + d2 * d2 + d3 * d3, sbuf);
    rs2 = rsqrtf(q / C + LN_EPS);
    p0 = (pv.x - mu2) * rs2 * gv.x + bv.x;
    p1 = (pv.y - mu2) * rs2 * gv.y + bv.y;
    p2 = (pv.z - mu2) * rs2 * gv.z + bv.z;
    p3 = (pv.w - mu2) * rs2 * gv.w + bv.w;
  }

  size_t o = (size_t)m * C + c0;
  {
    float4 mk = *(const float4*)(mx0 + c0);
    ushort4 ov;
    ov.x = f2bf(h0 * mk.x + p0 * (1.f - mk.x));
    ov.y = f2bf(h1 * mk.y + p1 * (1.f - mk.y));
    ov.z = f2bf(h2 * mk.z + p2 * (1.f - mk.z));
    ov.w = f2bf(h3 * mk.w + p3 * (1.f - mk.w));
    *(ushort4*)(o0 + o) = ov;
  }
  {
    float4 mk = *(const float4*)(mx1 + c0);
    ushort4 ov;
    ov.x = f2bf(h0 * mk.x + p0 * (1.f - mk.x));
    ov.y = f2bf(h1 * mk.y + p1 * (1.f - mk.y));
    ov.z = f2bf(h2 * mk.z + p2 * (1.f - mk.z));
    ov.w = f2bf(h3 * mk.w + p3 * (1.f - mk.w));
    *(ushort4*)(o1 + o) = ov;
  }
  if (NOUT > 2) {
    float4 mk = *(const float4*)(mx2 + c0);
    ushort4 ov;
    ov.x = f2bf(h0 * mk.x + p0 * (1.f - mk.x));
    ov.y = f2bf(h1 * mk.y + p1 * (1.f - mk.y));
    ov.z = f2bf(h2 * mk.z + p2 * (1.f - mk.z));
    ov.w = f2bf(h3 * mk.w + p3 * (1.f - mk.w));
    *(ushort4*)(o2 + o) = ov;
  }
}

// ---------------------------------------------------------------------------
// bf16 WMMA GEMM: out[M,N] = A[M,K] @ W^T where Wt is [N,K] (pre-transposed).
// Block tile 128x128, BK=32, 8 waves; wave tile 32x64 = 2x4 WMMA.
// EPI: 0 = f32 store, 1 = sigmoid f32, 2 = relu^2 -> bf16.
// ---------------------------------------------------------------------------
__device__ __forceinline__ v16bf frag_a_ld(const unsigned short* s, int mrow,
                                           int lane) {
  const unsigned short* p = s + (mrow + (lane & 15)) * LDK;
  int kh = (lane >> 4) << 3;  // 0 or 8
  FragBF f;
#pragma unroll
  for (int j = 0; j < 4; ++j) {
    f.u[j]     = *(const unsigned*)(p + kh + 2 * j);
    f.u[j + 4] = *(const unsigned*)(p + 16 + kh + 2 * j);
  }
  return f.v;
}

__device__ __forceinline__ v16bf frag_b_ld(const unsigned short* s, int ncol,
                                           int lane) {
  const unsigned short* p = s + (ncol + (lane & 15)) * LDK;
  int kh = (lane >> 4) << 3;  // 0 or 8
  FragBF f;
#pragma unroll
  for (int j = 0; j < 4; ++j) {
    f.u[j]     = *(const unsigned*)(p + kh + 2 * j);
    f.u[j + 4] = *(const unsigned*)(p + 16 + kh + 2 * j);
  }
  return f.v;
}

template <int EPI>
__global__ __launch_bounds__(256) void k_gemm_bf16(
    const unsigned short* __restrict__ A, const unsigned short* __restrict__ Wt,
    void* __restrict__ outp, int M, int N, int K) {
  __shared__ unsigned short sA[2][128 * LDK];
  __shared__ unsigned short sB[2][128 * LDK];
  const int tid = threadIdx.x, lane = tid & 31, wave = tid >> 5;
  const int bm = blockIdx.y, bn = blockIdx.x;
  const int wm = (wave >> 1) * 32;  // 0,32,64,96
  const int wn = (wave & 1) * 64;   // 0,64

  const unsigned short* Ab = A + (size_t)bm * 128 * K;
  const unsigned short* Bb = Wt + (size_t)bn * 128 * K;
  const int r0 = tid >> 2;       // 0..63
  const int ck = (tid & 3) * 8;  // 16B chunk within a 64B tile row

  v8f acc[2][4] = {};

  auto stage = [&](int buf, int k0) {
#pragma unroll
    for (int i = 0; i < 2; ++i) {
      int r = r0 + i * 64;
      async_copy16(Ab + (size_t)r * K + k0 + ck, &sA[buf][r * LDK + ck]);
#if !HAVE_TDM
      async_copy16(Bb + (size_t)r * K + k0 + ck, &sB[buf][r * LDK + ck]);
#endif
    }
#if HAVE_TDM
    if (wave == 0) {  // one TDM descriptor moves the whole 128x64B weight tile
      tdm_load_2d((unsigned)(unsigned long long)&sB[buf][0], Bb + k0,
                  /*tile_x=*/32, /*tile_y=*/128, /*stride=*/K,
                  /*tensor_x=*/K, /*tensor_y=*/128);
    }
#endif
  };

  auto wait_stage = [&]() {
    wait_async0();
#if HAVE_TDM
    if (wave == 0) __builtin_amdgcn_s_wait_tensorcnt(0);
#endif
    __syncthreads();  // publish DMA results to all waves
  };

  stage(0, 0);
  wait_stage();

  const int nk = K >> 5;
  for (int kt = 0; kt < nk; ++kt) {
    const int cur = kt & 1;
    if (kt + 1 < nk) stage(cur ^ 1, (kt + 1) << 5);  // DMA next tile now

    const unsigned short* pA = sA[cur];
    const unsigned short* pB = sB[cur];
    v16bf a0 = frag_a_ld(pA, wm, lane);
    v16bf a1 = frag_a_ld(pA, wm + 16, lane);
    v16bf b0 = frag_b_ld(pB, wn, lane);
    v16bf b1 = frag_b_ld(pB, wn + 16, lane);
    v16bf b2 = frag_b_ld(pB, wn + 32, lane);
    v16bf b3 = frag_b_ld(pB, wn + 48, lane);
    acc[0][0] = wmma_bf16(a0, b0, acc[0][0]);
    acc[0][1] = wmma_bf16(a0, b1, acc[0][1]);
    acc[0][2] = wmma_bf16(a0, b2, acc[0][2]);
    acc[0][3] = wmma_bf16(a0, b3, acc[0][3]);
    acc[1][0] = wmma_bf16(a1, b0, acc[1][0]);
    acc[1][1] = wmma_bf16(a1, b1, acc[1][1]);
    acc[1][2] = wmma_bf16(a1, b2, acc[1][2]);
    acc[1][3] = wmma_bf16(a1, b3, acc[1][3]);

    wait_stage();
  }

  // Store. C/D layout: VGPR r -> M=r (lanes 0-15) / M=r+8 (lanes 16-31), N=lane&15.
  const int l15 = lane & 15, lhi = (lane >> 4) * 8;
  float* outF = (float*)outp;
  unsigned short* outH = (unsigned short*)outp;
#pragma unroll
  for (int i = 0; i < 2; ++i) {
#pragma unroll
    for (int j = 0; j < 4; ++j) {
      FragF f;
      f.v = acc[i][j];
      int gn  = bn * 128 + wn + j * 16 + l15;
      int gm0 = bm * 128 + wm + i * 16 + lhi;
#pragma unroll
      for (int r = 0; r < 8; ++r) {
        size_t o = (size_t)(gm0 + r) * N + gn;
        float v = f.f[r];
        if (EPI == 0) {
          outF[o] = v;
        } else if (EPI == 1) {
          outF[o] = 1.f / (1.f + __expf(-v));
        } else {
          float t = fmaxf(v, 0.f);
          outH[o] = f2bf(t * t);
        }
      }
    }
  }
}

// ---------------------------------------------------------------------------
// Numerically-stable WKV scan: 2 independent chains per thread (batches b and
// b+B/2, same channel) so each wave carries 2 TRANS dependency chains.
// y may alias k (read-then-write at the same t).
// ---------------------------------------------------------------------------
__global__ void k_wkv2(const float* __restrict__ w_,
                       const float* __restrict__ u_,
                       const float* __restrict__ k, const float* __restrict__ v,
                       float* __restrict__ y, int B, int T, int C) {
  int idx = blockIdx.x * blockDim.x + threadIdx.x;
  int half = (B >> 1) * C;
  if (idx >= half) return;
  int b0 = idx / C, c = idx % C;
  int b1 = b0 + (B >> 1);
  float w = w_[c], u = u_[c];
  size_t s0 = (size_t)b0 * T * C + c;
  size_t s1 = (size_t)b1 * T * C + c;
  float aa0 = 0.f, bb0 = 0.f, pp0 = NEG_INF;
  float aa1 = 0.f, bb1 = 0.f, pp1 = NEG_INF;
  for (int t = 0; t < T; ++t) {
    size_t o = (size_t)t * C;
    float k0 = k[s0 + o], v0 = v[s0 + o];
    float k1 = k[s1 + o], v1 = v[s1 + o];

    float q0 = fmaxf(pp0, u + k0);
    float q1 = fmaxf(pp1, u + k1);
    float e10 = __expf(pp0 - q0), e20 = __expf(u + k0 - q0);
    float e11 = __expf(pp1 - q1), e21 = __expf(u + k1 - q1);
    float y0 = (e10 * aa0 + e20 * v0) / (e10 + e20);
    float y1 = (e11 * aa1 + e21 * v1) / (e11 + e21);

    float r0 = fmaxf(pp0 + w, w + k0);
    float r1 = fmaxf(pp1 + w, w + k1);
    float f10 = __expf(pp0 + w - r0), f20 = __expf(w + k0 - r0);
    float f11 = __expf(pp1 + w - r1), f21 = __expf(w + k1 - r1);
    aa0 = f10 * aa0 + f20 * v0;
    aa1 = f11 * aa1 + f21 * v1;
    bb0 = f10 + f20;
    bb1 = f11 + f21;
    pp0 = r0 + __logf(bb0);
    pp1 = r1 + __logf(bb1);

    y[s0 + o] = y0;
    y[s1 + o] = y1;
  }
}

// out[i] = base[i] + g[i] * y[i]   (float4 vectorized; base may alias out)
__global__ void k_gate_add(const float* __restrict__ base,
                           const float* __restrict__ g,
                           const float* __restrict__ y, float* __restrict__ out,
                           int n) {
  int i = (blockIdx.x * blockDim.x + threadIdx.x) * 4;
  if (i + 3 < n) {
    float4 b  = *(const float4*)(base + i);
    float4 gg = *(const float4*)(g + i);
    float4 yy = *(const float4*)(y + i);
    float4 o;
    o.x = b.x + gg.x * yy.x;
    o.y = b.y + gg.y * yy.y;
    o.z = b.z + gg.z * yy.z;
    o.w = b.w + gg.w * yy.w;
    *(float4*)(out + i) = o;
  }
}

// ---------------------------------------------------------------------------
extern "C" void kernel_launch(void* const* d_in, const int* in_sizes, int n_in,
                              void* d_out, int out_size, void* d_ws,
                              size_t ws_size, hipStream_t stream) {
  const float* x   = (const float*)d_in[0];
  const float* td  = (const float*)d_in[1];   // time_decay
  const float* tf  = (const float*)d_in[2];   // time_first
  const float* wtk = (const float*)d_in[3];
  const float* wtv = (const float*)d_in[4];
  const float* wtr = (const float*)d_in[5];
  const float* wck = (const float*)d_in[6];
  const float* wcv = (const float*)d_in[7];
  const float* wcr = (const float*)d_in[8];
  const float* l1g = (const float*)d_in[9];
  const float* l1b = (const float*)d_in[10];
  const float* l2g = (const float*)d_in[11];
  const float* l2b = (const float*)d_in[12];
  const float* mk  = (const float*)d_in[13];
  const float* mv  = (const float*)d_in[14];
  const float* mr  = (const float*)d_in[15];
  const float* cmk = (const float*)d_in[16];
  const float* cmr = (const float*)d_in[17];

  const int C = in_sizes[1];          // 1024
  const int H = in_sizes[6] / C;      // 4096
  const int T = 2048;                 // reference shape
  const int B = in_sizes[0] / (T * C);
  const int M = B * T;                // 16384

  // Workspace bump allocator (256B aligned).
  char* wsp = (char*)d_ws;
  auto alloc = [&](size_t bytes) -> void* {
    void* p = (void*)wsp;
    wsp += (bytes + 255) & ~(size_t)255;
    return p;
  };
  // Transposed bf16 weights [N,K]:
  unsigned short* wtk_b = (unsigned short*)alloc((size_t)C * C * 2);
  unsigned short* wtv_b = (unsigned short*)alloc((size_t)C * C * 2);
  unsigned short* wtr_b = (unsigned short*)alloc((size_t)C * C * 2);
  unsigned short* wcr_b = (unsigned short*)alloc((size_t)C * C * 2);
  unsigned short* wck_b = (unsigned short*)alloc((size_t)C * H * 2);  // [H,C]
  unsigned short* wcv_b = (unsigned short*)alloc((size_t)H * C * 2);  // [C,H]
  unsigned short* xk_b  = (unsigned short*)alloc((size_t)M * C * 2);
  unsigned short* xv_b  = (unsigned short*)alloc((size_t)M * C * 2);
  unsigned short* xr_b  = (unsigned short*)alloc((size_t)M * C * 2);
  float* kf = (float*)alloc((size_t)M * C * 4);  // later aliased: y, then crs
  float* vf = (float*)alloc((size_t)M * C * 4);
  float* rf = (float*)alloc((size_t)M * C * 4);  // later aliased: cv
  unsigned short* ck_b = (unsigned short*)alloc((size_t)M * C * 2);
  unsigned short* cr_b = (unsigned short*)alloc((size_t)M * C * 2);
  unsigned short* kk_b = (unsigned short*)alloc((size_t)M * H * 2);

  float* x1   = (float*)d_out;  // residual after time-mixing lives in d_out
  float* yf   = kf;             // WKV output aliases k (read-then-write per t)
  float* cvf  = rf;             // r consumed before channel GEMMs
  float* crsf = kf;             // k/y consumed before cr GEMM

  const int blk = 256;
  auto convT = [&](const float* src, unsigned short* dst, int Kd, int Nd) {
    k_convert_t<<<dim3(Nd / 32, Kd / 32), blk, 0, stream>>>(src, dst, Kd, Nd);
  };

  // 1) weights -> bf16, transposed to [N,K]
  convT(wtk, wtk_b, C, C);
  convT(wtv, wtv_b, C, C);
  convT(wtr, wtr_b, C, C);
  convT(wcr, wcr_b, C, C);
  convT(wck, wck_b, C, H);
  convT(wcv, wcv_b, H, C);

  // 2) LN1 + shift + mix -> xk, xv, xr (bf16)
  k_ln_shift_mix<3><<<dim3(M), blk, 0, stream>>>(x, l1g, l1b, mk, mv, mr, xk_b,
                                                 xv_b, xr_b, T, C);

  // 3) time-mixing GEMMs
  dim3 gCC(C / 128, M / 128);
  k_gemm_bf16<0><<<gCC, blk, 0, stream>>>(xk_b, wtk_b, (void*)kf, M, C, C);
  k_gemm_bf16<0><<<gCC, blk, 0, stream>>>(xv_b, wtv_b, (void*)vf, M, C, C);
  k_gemm_bf16<1><<<gCC, blk, 0, stream>>>(xr_b, wtr_b, (void*)rf, M, C, C);

  // 4) WKV scan (serial over T; 2 chains per thread)
  k_wkv2<<<dim3(((B / 2) * C + blk - 1) / blk), blk, 0, stream>>>(td, tf, kf,
                                                                  vf, yf, B, T,
                                                                  C);

  // 5) x1 = x + sigmoid(r) * y
  k_gate_add<<<dim3((M * C / 4 + blk - 1) / blk), blk, 0, stream>>>(x, rf, yf,
                                                                    x1, M * C);

  // 6) LN2 + shift + mix -> ck, cr (bf16)
  k_ln_shift_mix<2><<<dim3(M), blk, 0, stream>>>(x1, l2g, l2b, cmk, cmr,
                                                 nullptr, ck_b, cr_b, nullptr,
                                                 T, C);

  // 7) channel-mixing GEMMs
  dim3 gCH(H / 128, M / 128);
  k_gemm_bf16<2><<<gCH, blk, 0, stream>>>(ck_b, wck_b, (void*)kk_b, M, H, C);
  k_gemm_bf16<0><<<gCC, blk, 0, stream>>>(kk_b, wcv_b, (void*)cvf, M, C, H);
  k_gemm_bf16<1><<<gCC, blk, 0, stream>>>(cr_b, wcr_b, (void*)crsf, M, C, C);

  // 8) out = x1 + sigmoid(cr) * cv   (in place on d_out)
  k_gate_add<<<dim3((M * C / 4 + blk - 1) / blk), blk, 0, stream>>>(
      x1, crsf, cvf, x1, M * C);
}
